// I2GNN_69028714381396
// MI455X (gfx1250) — matrix operand order repaired and checked by reading, not compiled, not run on hardware
//
#include <hip/hip_runtime.h>
#include <hip/hip_bf16.h>
#include <math.h>

#define NNODES 160000
#define NEDGES 2560000
#define NS2    16000
#define NS1    1600
#define NG     160
#define NL     5
#define HD     32
#define NZV    50
#define BN_EPS 1e-5f

typedef __attribute__((ext_vector_type(16))) __bf16 v16bf;
typedef __attribute__((ext_vector_type(8)))  __bf16 v8bf;
typedef __attribute__((ext_vector_type(8)))  float  v8f;

// ---------------------------------------------------------------- WMMA helpers

__device__ __forceinline__ v8f wmma_bf16(v16bf a, v16bf b, v8f c) {
  // D = A(16x32 bf16) * B(32x16 bf16) + C(16x16 f32)
  return __builtin_amdgcn_wmma_f32_16x16x32_bf16(false, a, false, b, (short)0, c,
                                                 false, false);
}

// A fragment from a row-major [*,32] f32 matrix (converted data: agg / pooled bufs).
// 16-bit A layout (ISA 7.12.2): lanes0-15 K-halves {0-7,16-23}, lanes16-31 {8-15,24-31}.
__device__ __forceinline__ v16bf loadA_f32(const float* __restrict__ X, long row0, int lane) {
  const float* xr = X + (row0 + (lane & 15)) * HD;
  int h8 = ((lane >> 4) & 1) * 8;
  v16bf a;
#pragma unroll
  for (int i = 0; i < 8; ++i) {
    a[i]     = (__bf16)xr[h8 + i];
    a[i + 8] = (__bf16)xr[16 + h8 + i];
  }
  return a;
}

// A fragment from a row-major [*,32] bf16 mirror: two aligned 16B chunks, no converts.
__device__ __forceinline__ v16bf loadA_bf16(const __bf16* __restrict__ X, long row0, int lane) {
  const __bf16* xr = X + (row0 + (lane & 15)) * HD;
  int h8 = ((lane >> 4) & 1) * 8;
  v8bf lo = *(const v8bf*)(xr + h8);
  v8bf hi = *(const v8bf*)(xr + 16 + h8);
  v16bf a;
#pragma unroll
  for (int i = 0; i < 8; ++i) { a[i] = lo[i]; a[i + 8] = hi[i]; }
  return a;
}

// same, from a per-wave LDS 16x32 bf16 tile
__device__ __forceinline__ v16bf loadA_lds(const __bf16* tile, int lane) {
  const __bf16* xr = tile + (lane & 15) * HD;
  int h8 = ((lane >> 4) & 1) * 8;
  v8bf lo = *(const v8bf*)(xr + h8);
  v8bf hi = *(const v8bf*)(xr + 16 + h8);
  v16bf a;
#pragma unroll
  for (int i = 0; i < 8; ++i) { a[i] = lo[i]; a[i + 8] = hi[i]; }
  return a;
}

// B fragment: K=32 x N=16 tile of a row-major [K, ld] bf16 weight matrix.
// lane supplies column N = n0+(lane&15); K-halves {0-7,16-23} (lanes0-15), {8-15,24-31} (lanes16-31).
__device__ __forceinline__ v16bf loadB16(const __bf16* __restrict__ W, int ld, int k0,
                                         int n0, int lane) {
  int n = n0 + (lane & 15);
  int h8 = ((lane >> 4) & 1) * 8;
  v16bf b;
#pragma unroll
  for (int i = 0; i < 8; ++i) {
    b[i]     = W[(long)(k0 + h8 + i) * ld + n];
    b[i + 8] = W[(long)(k0 + 16 + h8 + i) * ld + n];
  }
  return b;
}

// C init with a per-column bias (column N = n0 + lane&15, same for all 8 rows)
__device__ __forceinline__ v8f biasC(const float* __restrict__ b, int n0, int lane) {
  float v = b[n0 + (lane & 15)];
  v8f c;
#pragma unroll
  for (int i = 0; i < 8; ++i) c[i] = v;
  return c;
}

// D store: lane holds col = n0+(lane&15), rows M = 8*(lane>>4)+i
__device__ __forceinline__ void storeD(float* __restrict__ X, long row0, int n0,
                                       int lane, v8f c) {
  int col = n0 + (lane & 15);
  long rb = row0 + ((lane >> 4) << 3);
#pragma unroll
  for (int i = 0; i < 8; ++i) X[(rb + i) * HD + col] = c[i];
}

__device__ __forceinline__ void atomAddF(float* p, float v) {
  unsafeAtomicAdd(p, v);   // global_atomic_add_f32 (L2 atomic)
}

// ---------------------------------------------------------------- kernels

__global__ void zero_kernel(float* __restrict__ p, long n) {
  long i = (long)blockIdx.x * blockDim.x + threadIdx.x;
  if (i < n) p[i] = 0.f;
}

__global__ void cvt_bf16_kernel(const float* __restrict__ s, __bf16* __restrict__ d, int n) {
  int i = blockIdx.x * blockDim.x + threadIdx.x;
  if (i < n) d[i] = (__bf16)s[i];
}

// x = (l==0) ? zl : concat(x_prev, zl) @ tw + tb ;  m = x @ conv_w
// 256 threads = 8 waves, each wave owns a 16-node tile (N % 128 == 0).
__global__ __launch_bounds__(256)
void layer_front_kernel(const int* __restrict__ z, const float* __restrict__ emb_l,
                        const __bf16* __restrict__ tw_l, const float* __restrict__ tb_l,
                        const __bf16* __restrict__ conv_l,
                        const __bf16* __restrict__ xbf_in, float* __restrict__ x_out,
                        __bf16* __restrict__ xbf_out, float* __restrict__ m_out,
                        int has_prev) {
  __shared__ __bf16 tile[8][16 * HD];
  int lane = threadIdx.x & 31;
  int wave = threadIdx.x >> 5;
  long row0 = ((long)blockIdx.x * 8 + wave) * 16;
  int r  = (int)row0 + (lane & 15);
  int h8 = ((lane >> 4) & 1) * 8;

  // embedding sum -> A-layout f32 values for this lane
  int z0 = z[2 * r], z1 = z[2 * r + 1];
  const float* e0 = emb_l + (long)z0 * HD;
  const float* e1 = emb_l + (long)z1 * HD;
  float az[16];
#pragma unroll
  for (int i = 0; i < 8; ++i) {
    az[i]     = e0[h8 + i]      + e1[h8 + i];
    az[i + 8] = e0[16 + h8 + i] + e1[16 + h8 + i];
  }
  v16bf a_z;
#pragma unroll
  for (int i = 0; i < 16; ++i) a_z[i] = (__bf16)az[i];

  v16bf a_x;
  if (has_prev) {
    v16bf a_prev = loadA_bf16(xbf_in, row0, lane);   // mirror from bn_apply
#pragma unroll
    for (int t = 0; t < 2; ++t) {
      int n0 = t * 16;
      v8f c = biasC(tb_l, n0, lane);
      c = wmma_bf16(a_prev, loadB16(tw_l, HD, 0,  n0, lane), c);  // K 0..31 (x_prev)
      c = wmma_bf16(a_z,    loadB16(tw_l, HD, 32, n0, lane), c);  // K 32..63 (zl)
      storeD(x_out, row0, n0, lane, c);
      int col = n0 + (lane & 15);
      long rb = row0 + ((lane >> 4) << 3);
      int lrb = (lane >> 4) << 3;
#pragma unroll
      for (int i = 0; i < 8; ++i) {
        __bf16 bv = (__bf16)c[i];
        xbf_out[(rb + i) * HD + col]   = bv;     // global bf16 mirror
        tile[wave][(lrb + i) * HD + col] = bv;   // wave-local re-layout
      }
    }
    a_x = loadA_lds(&tile[wave][0], lane);       // wave-local RAW through LDS
  } else {
    float* xr  = x_out + (long)r * HD;           // x = zl
    __bf16* xb = xbf_out + (long)r * HD;
#pragma unroll
    for (int i = 0; i < 8; ++i) {
      xr[h8 + i]      = az[i];
      xr[16 + h8 + i] = az[i + 8];
      xb[h8 + i]      = a_z[i];
      xb[16 + h8 + i] = a_z[i + 8];
    }
    a_x = a_z;
  }

  // m = x @ conv_w
#pragma unroll
  for (int t = 0; t < 2; ++t) {
    int n0 = t * 16;
    v8f c = {0.f, 0.f, 0.f, 0.f, 0.f, 0.f, 0.f, 0.f};
    c = wmma_bf16(a_x, loadB16(conv_l, HD, 0, n0, lane), c);
    storeD(m_out, row0, n0, lane, c);
  }
}

// agg[dst] += m[src]; 8 lanes per edge, float4 per lane, hardware f32 atomics.
__global__ void edge_scatter_kernel(const int* __restrict__ src, const int* __restrict__ dst,
                                    const float* __restrict__ m, float* __restrict__ agg) {
  long idx = (long)blockIdx.x * blockDim.x + threadIdx.x;
  if (idx >= (long)NEDGES * 8) return;
  int e = (int)(idx >> 3), q = (int)(idx & 7);
  int s = src[e], d = dst[e];
  const float4 v = *(const float4*)(m + (long)s * HD + q * 4);
  float* ap = agg + (long)d * HD + q * 4;
  atomAddF(ap + 0, v.x);
  atomAddF(ap + 1, v.y);
  atomAddF(ap + 2, v.z);
  atomAddF(ap + 3, v.w);
}

// GRU cell (12 WMMAs per 16-node tile) + in-place x update + BN statistics.
__global__ __launch_bounds__(256)
void gru_bn_kernel(const float* __restrict__ agg, float* __restrict__ x,
                   const __bf16* __restrict__ xbf,
                   const __bf16* __restrict__ wi, const __bf16* __restrict__ wh,
                   const float* __restrict__ bi, const float* __restrict__ bh,
                   float* __restrict__ stats) {
  __shared__ float ssum[HD], ssq[HD];
  if (threadIdx.x < HD) { ssum[threadIdx.x] = 0.f; ssq[threadIdx.x] = 0.f; }
  __syncthreads();

  int lane = threadIdx.x & 31, wave = threadIdx.x >> 5;
  long row0 = ((long)blockIdx.x * 8 + wave) * 16;
  v16bf a_agg = loadA_f32(agg, row0, lane);    // atomically-built, f32 only
  v16bf a_x   = loadA_bf16(xbf, row0, lane);   // mirror from layer_front

  v8f gi[6], gh[6];
#pragma unroll
  for (int t = 0; t < 6; ++t) {
    gi[t] = wmma_bf16(a_agg, loadB16(wi, 3 * HD, 0, t * 16, lane), biasC(bi, t * 16, lane));
    gh[t] = wmma_bf16(a_x,   loadB16(wh, 3 * HD, 0, t * 16, lane), biasC(bh, t * 16, lane));
  }

  long rb = row0 + ((lane >> 4) << 3);
#pragma unroll
  for (int g = 0; g < 2; ++g) {
    int c = g * 16 + (lane & 15);
    float lsum = 0.f, lsq = 0.f;
#pragma unroll
    for (int i = 0; i < 8; ++i) {
      float ir = gi[g][i], iz = gi[2 + g][i], inn = gi[4 + g][i];
      float hr = gh[g][i], hz = gh[2 + g][i], hn  = gh[4 + g][i];
      float xv = x[(rb + i) * HD + c];
      float rr = 1.f / (1.f + __expf(-(ir + hr)));
      float zg = 1.f / (1.f + __expf(-(iz + hz)));
      float nn = tanhf(inn + rr * hn);
      float xo = (1.f - zg) * nn + zg * xv;
      x[(rb + i) * HD + c] = xo;
      lsum += xo;
      lsq  += xo * xo;
    }
    atomicAdd(&ssum[c], lsum);   // ds_add_f32
    atomicAdd(&ssq[c],  lsq);
  }
  __syncthreads();
  if (threadIdx.x < HD) {
    atomAddF(&stats[threadIdx.x],      ssum[threadIdx.x]);
    atomAddF(&stats[HD + threadIdx.x], ssq[threadIdx.x]);
  }
}

__global__ void bn_apply_kernel(float* __restrict__ x, __bf16* __restrict__ xbf,
                                const float* __restrict__ stats,
                                const float* __restrict__ g, const float* __restrict__ b) {
  long idx = (long)blockIdx.x * blockDim.x + threadIdx.x;
  if (idx >= (long)NNODES * HD) return;
  int c    = (int)(idx & (HD - 1));
  float mu = stats[c] * (1.f / NNODES);
  float vr = stats[HD + c] * (1.f / NNODES) - mu * mu;
  float v  = (x[idx] - mu) * rsqrtf(vr + BN_EPS) * g[c] + b[c];
  x[idx]   = v;
  xbf[idx] = (__bf16)v;   // refresh mirror for next layer's A fragments
}

// segment-sum pooling: out[map[r]] += in[r]
__global__ void pool_scatter_kernel(const float* __restrict__ in, const int* __restrict__ map,
                                    float* __restrict__ out, int rows) {
  long idx = (long)blockIdx.x * blockDim.x + threadIdx.x;
  if (idx >= (long)rows * 8) return;
  int r = (int)(idx >> 3), q = (int)(idx & 7);
  int s = map[r];
  const float4 v = *(const float4*)(in + (long)r * HD + q * 4);
  float* op = out + (long)s * HD + q * 4;
  atomAddF(op + 0, v.x);
  atomAddF(op + 1, v.y);
  atomAddF(op + 2, v.z);
  atomAddF(op + 3, v.w);
}

// out = relu(in @ w1 + b1) @ w2 + b2, rows in 16-row tiles (one wave each)
__global__ __launch_bounds__(256)
void mlp_kernel(const float* __restrict__ in, float* __restrict__ out,
                const __bf16* __restrict__ w1, const float* __restrict__ b1,
                const __bf16* __restrict__ w2, const float* __restrict__ b2, int ntiles) {
  __shared__ __bf16 tile[8][16 * HD];
  int lane = threadIdx.x & 31, wave = threadIdx.x >> 5;
  int t = blockIdx.x * 8 + wave;
  if (t >= ntiles) return;                     // wave-uniform guard, EXEC stays full
  long row0 = (long)t * 16;
  v16bf a = loadA_f32(in, row0, lane);         // pooled input is f32 (atomics)
#pragma unroll
  for (int s = 0; s < 2; ++s) {
    int n0 = s * 16;
    v8f c = wmma_bf16(a, loadB16(w1, HD, 0, n0, lane), biasC(b1, n0, lane));
    int col = n0 + (lane & 15), rb = (lane >> 4) << 3;
#pragma unroll
    for (int i = 0; i < 8; ++i)
      tile[wave][(rb + i) * HD + col] = (__bf16)fmaxf(c[i], 0.f);
  }
  v16bf ah = loadA_lds(&tile[wave][0], lane);
#pragma unroll
  for (int s = 0; s < 2; ++s) {
    int n0 = s * 16;
    v8f c = wmma_bf16(ah, loadB16(w2, HD, 0, n0, lane), biasC(b2, n0, lane));
    storeD(out, row0, n0, lane, c);
  }
}

// head: fc1 + ELU + BN(no affine) + fc3 over [160,*] — tiny, one block
__global__ __launch_bounds__(256)
void head_kernel(const float* __restrict__ gb, const float* __restrict__ fc1w,
                 const float* __restrict__ fc1b, const float* __restrict__ fc3w,
                 const float* __restrict__ fc3b, float* __restrict__ out) {
  __shared__ float t[NG * 16];
  __shared__ float mu[16], rs[16];
  for (int idx = threadIdx.x; idx < NG * 16; idx += 256) {
    int r = idx >> 4, c = idx & 15;
    float s = fc1b[c];
#pragma unroll
    for (int k = 0; k < HD; ++k) s += gb[r * HD + k] * fc1w[k * 16 + c];
    t[idx] = s > 0.f ? s : (__expf(s) - 1.f);          // ELU
  }
  __syncthreads();
  if (threadIdx.x < 16) {
    int c = threadIdx.x;
    float su = 0.f, sq = 0.f;
    for (int r = 0; r < NG; ++r) { float v = t[r * 16 + c]; su += v; sq += v * v; }
    float m = su * (1.f / NG);
    mu[c] = m;
    rs[c] = rsqrtf(sq * (1.f / NG) - m * m + BN_EPS);
  }
  __syncthreads();
  for (int idx = threadIdx.x; idx < NG * 16; idx += 256) {
    int r = idx >> 4, c = idx & 15;
    float s = fc3b[c];
#pragma unroll
    for (int k = 0; k < 16; ++k) s += (t[r * 16 + k] - mu[k]) * rs[k] * fc3w[k * 16 + c];
    out[idx] = s;
  }
}

// ---------------------------------------------------------------- launch

extern "C" void kernel_launch(void* const* d_in, const int* in_sizes, int n_in,
                              void* d_out, int out_size, void* d_ws, size_t ws_size,
                              hipStream_t stream) {
  (void)in_sizes; (void)n_in; (void)out_size; (void)ws_size;
  const int*   z     = (const int*)d_in[0];
  const int*   ei    = (const int*)d_in[1];
  const int*   n2s2  = (const int*)d_in[2];
  const int*   s2s1  = (const int*)d_in[3];
  const int*   s1g   = (const int*)d_in[4];
  const float* emb   = (const float*)d_in[5];
  const float* tw    = (const float*)d_in[6];
  const float* tb    = (const float*)d_in[7];
  const float* convw = (const float*)d_in[8];
  const float* gwi   = (const float*)d_in[9];
  const float* gwh   = (const float*)d_in[10];
  const float* gbi   = (const float*)d_in[11];
  const float* gbh   = (const float*)d_in[12];
  const float* bng   = (const float*)d_in[13];
  const float* bnb   = (const float*)d_in[14];
  const float* ew1   = (const float*)d_in[15];
  const float* eb1   = (const float*)d_in[16];
  const float* ew2   = (const float*)d_in[17];
  const float* eb2   = (const float*)d_in[18];
  const float* nw1   = (const float*)d_in[19];
  const float* nb1   = (const float*)d_in[20];
  const float* nw2   = (const float*)d_in[21];
  const float* nb2   = (const float*)d_in[22];
  const float* f1w   = (const float*)d_in[23];
  const float* f1b   = (const float*)d_in[24];
  const float* f3w   = (const float*)d_in[25];
  const float* f3b   = (const float*)d_in[26];
  const int* src = ei;
  const int* dst = ei + NEDGES;

  // workspace carve-out (256B aligned regions)
  char* wp = (char*)d_ws;
  auto carve = [&](size_t bytes) -> void* {
    void* p = (void*)wp;
    wp += (bytes + 255) & ~(size_t)255;
    return p;
  };
  float*  x     = (float*)carve((size_t)NNODES * HD * 4);
  float*  m     = (float*)carve((size_t)NNODES * HD * 4);
  float*  agg   = (float*)carve((size_t)NNODES * HD * 4);
  float*  stats = (float*)carve(2 * HD * 4);
  float*  s2a   = (float*)carve((size_t)NS2 * HD * 4);
  float*  s2b   = (float*)carve((size_t)NS2 * HD * 4);
  float*  s1a   = (float*)carve((size_t)NS1 * HD * 4);
  float*  s1b   = (float*)carve((size_t)NS1 * HD * 4);
  float*  gb    = (float*)carve((size_t)NG * HD * 4);
  __bf16* xbf   = (__bf16*)carve((size_t)NNODES * HD * 2);
  __bf16* twb   = (__bf16*)carve((size_t)NL * 2 * HD * HD * 2);
  __bf16* convb = (__bf16*)carve((size_t)NL * HD * HD * 2);
  __bf16* gwib  = (__bf16*)carve((size_t)NL * HD * 3 * HD * 2);
  __bf16* gwhb  = (__bf16*)carve((size_t)NL * HD * 3 * HD * 2);
  __bf16* ew1b  = (__bf16*)carve((size_t)HD * HD * 2);
  __bf16* ew2b  = (__bf16*)carve((size_t)HD * HD * 2);
  __bf16* nw1b  = (__bf16*)carve((size_t)HD * HD * 2);
  __bf16* nw2b  = (__bf16*)carve((size_t)HD * HD * 2);

  // pre-convert all WMMA weights to bf16 (tiny; keeps B-operand loads packed)
  auto cvt = [&](const float* s, __bf16* d, int n) {
    cvt_bf16_kernel<<<(n + 255) / 256, 256, 0, stream>>>(s, d, n);
  };
  cvt(tw,    twb,   NL * 2 * HD * HD);
  cvt(convw, convb, NL * HD * HD);
  cvt(gwi,   gwib,  NL * HD * 3 * HD);
  cvt(gwh,   gwhb,  NL * HD * 3 * HD);
  cvt(ew1,   ew1b,  HD * HD);
  cvt(ew2,   ew2b,  HD * HD);
  cvt(nw1,   nw1b,  HD * HD);
  cvt(nw2,   nw2b,  HD * HD);

  // zero pooled accumulators (fresh every call — graph-replay safe)
  zero_kernel<<<(NS2 * HD + 255) / 256, 256, 0, stream>>>(s2a, (long)NS2 * HD);
  zero_kernel<<<(NS1 * HD + 255) / 256, 256, 0, stream>>>(s1a, (long)NS1 * HD);
  zero_kernel<<<(NG * HD + 255) / 256, 256, 0, stream>>>(gb, (long)NG * HD);

  for (int l = 0; l < NL; ++l) {
    layer_front_kernel<<<NNODES / 128, 256, 0, stream>>>(
        z, emb + (size_t)l * NZV * HD, twb + (size_t)l * 2 * HD * HD, tb + l * HD,
        convb + (size_t)l * HD * HD, xbf, x, xbf, m, l > 0);
    zero_kernel<<<(NNODES * HD + 255) / 256, 256, 0, stream>>>(agg, (long)NNODES * HD);
    zero_kernel<<<1, 64, 0, stream>>>(stats, 2 * HD);
    edge_scatter_kernel<<<(int)(((long)NEDGES * 8 + 255) / 256), 256, 0, stream>>>(
        src, dst, m, agg);
    gru_bn_kernel<<<NNODES / 128, 256, 0, stream>>>(
        agg, x, xbf, gwib + (size_t)l * HD * 3 * HD, gwhb + (size_t)l * HD * 3 * HD,
        gbi + l * 3 * HD, gbh + l * 3 * HD, stats);
    bn_apply_kernel<<<(NNODES * HD + 255) / 256, 256, 0, stream>>>(
        x, xbf, stats, bng + l * HD, bnb + l * HD);
  }

  pool_scatter_kernel<<<(NNODES * 8 + 255) / 256, 256, 0, stream>>>(x, n2s2, s2a, NNODES);
  mlp_kernel<<<(NS2 / 16 + 7) / 8, 256, 0, stream>>>(s2a, s2b, ew1b, eb1, ew2b, eb2, NS2 / 16);
  pool_scatter_kernel<<<(NS2 * 8 + 255) / 256, 256, 0, stream>>>(s2b, s2s1, s1a, NS2);
  mlp_kernel<<<(NS1 / 16 + 7) / 8, 256, 0, stream>>>(s1a, s1b, nw1b, nb1, nw2b, nb2, NS1 / 16);
  pool_scatter_kernel<<<(NS1 * 8 + 255) / 256, 256, 0, stream>>>(s1b, s1g, gb, NS1);
  head_kernel<<<1, 256, 0, stream>>>(gb, f1w, f1b, f3w, f3b, (float*)d_out);
}
